// NNUE_52063593562217
// MI455X (gfx1250) — compile-verified, matrix operand order checked by reading, not run on hardware
//
#include <hip/hip_runtime.h>
#include <hip/hip_fp16.h>

typedef __attribute__((ext_vector_type(16))) _Float16 v16h;
typedef __attribute__((ext_vector_type(8)))  _Float16 v8h;
typedef __attribute__((ext_vector_type(8)))  float    v8f;

#define HIDDEN 256
#define TWOH   512
#define L2N    32

union V16U { v16h v; v8h h[2]; };

__device__ __forceinline__ float clamp01(float x) {
    return fminf(fmaxf(x, 0.0f), 1.0f);
}

// ---------------------------------------------------------------------------
// Kernel 1: embedding-bag sum (white + black), bias, clamp, stm-select swap,
// write combined (B x 512) activations as f16 into workspace.
// One wave32 per bag; lane l owns columns [l*8, l*8+8).
// ---------------------------------------------------------------------------
__global__ __launch_bounds__(256) void nnue_embed_kernel(
    const int* __restrict__ widx, const int* __restrict__ woff,
    const int* __restrict__ bidx, const int* __restrict__ boff,
    const unsigned char* __restrict__ stm,
    const float* __restrict__ ftw, const float* __restrict__ ftb,
    _Float16* __restrict__ combined, int nbags, int wtot, int btot)
{
    int tid  = blockIdx.x * blockDim.x + threadIdx.x;
    int bag  = tid >> 5;
    int lane = tid & 31;
    if (bag >= nbags) return;

    int col = lane * 8;
    float aw[8] = {0.f,0.f,0.f,0.f,0.f,0.f,0.f,0.f};
    float ab[8] = {0.f,0.f,0.f,0.f,0.f,0.f,0.f,0.f};

    // white bag
    {
        int start = woff[bag];
        int end   = (bag + 1 < nbags) ? woff[bag + 1] : wtot;
        for (int base = start; base < end; base += 32) {
            int my  = (base + lane < end) ? widx[base + lane] : 0;
            int cnt = min(32, end - base);
            for (int f = 0; f < cnt; ++f) {
                int idx = __shfl(my, f, 32);
                const float4* row = (const float4*)(ftw + (size_t)idx * HIDDEN + col);
                float4 r0 = row[0], r1 = row[1];
                aw[0] += r0.x; aw[1] += r0.y; aw[2] += r0.z; aw[3] += r0.w;
                aw[4] += r1.x; aw[5] += r1.y; aw[6] += r1.z; aw[7] += r1.w;
            }
        }
    }
    // black bag
    {
        int start = boff[bag];
        int end   = (bag + 1 < nbags) ? boff[bag + 1] : btot;
        for (int base = start; base < end; base += 32) {
            int my  = (base + lane < end) ? bidx[base + lane] : 0;
            int cnt = min(32, end - base);
            for (int f = 0; f < cnt; ++f) {
                int idx = __shfl(my, f, 32);
                const float4* row = (const float4*)(ftw + (size_t)idx * HIDDEN + col);
                float4 r0 = row[0], r1 = row[1];
                ab[0] += r0.x; ab[1] += r0.y; ab[2] += r0.z; ab[3] += r0.w;
                ab[4] += r1.x; ab[5] += r1.y; ab[6] += r1.z; ab[7] += r1.w;
            }
        }
    }

    bool s = (stm[bag] != 0);
    v8h us, them;
    #pragma unroll
    for (int i = 0; i < 8; ++i) {
        float bias = ftb[col + i];
        float w = clamp01(aw[i] + bias);
        float b = clamp01(ab[i] + bias);
        us[i]   = (_Float16)(s ? b : w);
        them[i] = (_Float16)(s ? w : b);
    }
    _Float16* crow = combined + (size_t)bag * TWOH;
    *(v8h*)(crow + col)          = us;
    *(v8h*)(crow + HIDDEN + col) = them;
}

// ---------------------------------------------------------------------------
// Kernel 2: hidden = clip(combined @ l1_w^T + l1_b); out = hidden @ l2_w^T + l2_b
// One wave32 per 16-row tile: 16 K-steps x 2 N-tiles of v_wmma_f32_16x16x32_f16.
// l1_w converted to f16 in LDS once per block (32 KB).
// ---------------------------------------------------------------------------
__global__ __launch_bounds__(256) void nnue_head_kernel(
    const _Float16* __restrict__ combined,
    const float* __restrict__ l1w, const float* __restrict__ l1b,
    const float* __restrict__ l2w, const float* __restrict__ l2b,
    float* __restrict__ out, int nrows)
{
    __shared__ __align__(16) _Float16 w1h[L2N * TWOH]; // 32 KB of 320 KB WGP LDS
    for (int i = threadIdx.x; i < L2N * TWOH; i += 256)
        w1h[i] = (_Float16)l1w[i];
    __syncthreads();

    int wave    = blockIdx.x * (blockDim.x >> 5) + (threadIdx.x >> 5);
    int lane    = threadIdx.x & 31;
    int rowbase = wave * 16;
    if (rowbase >= nrows) return; // wave-uniform: active waves keep EXEC all-1s

    int m  = lane & 15;   // row within tile (A) / column N (B,C,D)
    int hi = lane >> 4;   // lane-half selects K sub-ranges per 16-bit WMMA layout
    int n  = m;

    const _Float16* arow  = combined + (size_t)(rowbase + m) * TWOH;
    const _Float16* b0row = &w1h[n * TWOH];
    const _Float16* b1row = &w1h[(n + 16) * TWOH];

    v8f c0 = {}; v8f c1 = {};
    #pragma unroll
    for (int kk = 0; kk < 16; ++kk) {
        int k0  = kk * 32;
        // 16-bit A layout: lanes 0-15 hold K = k0+0..7 and k0+16..23,
        //                  lanes 16-31 hold K = k0+8..15 and k0+24..31
        int kla = k0 + hi * 8;
        int klb = k0 + 16 + hi * 8;
        V16U a, b0, b1;
        a.h[0]  = *(const v8h*)(arow  + kla);
        a.h[1]  = *(const v8h*)(arow  + klb);
        b0.h[0] = *(const v8h*)(b0row + kla);
        b0.h[1] = *(const v8h*)(b0row + klb);
        b1.h[0] = *(const v8h*)(b1row + kla);
        b1.h[1] = *(const v8h*)(b1row + klb);
        c0 = __builtin_amdgcn_wmma_f32_16x16x32_f16(false, a.v, false, b0.v,
                                                    (short)0, c0, false, false);
        c1 = __builtin_amdgcn_wmma_f32_16x16x32_f16(false, a.v, false, b1.v,
                                                    (short)0, c1, false, false);
    }

    // Fused L1 bias/clip + L2 head. C/D layout: lane half holds N=lane&15,
    // VGPR r holds M = r (lanes 0-15) or r+8 (lanes 16-31).
    float bi0 = l1b[n], bi1 = l1b[n + 16];
    float w20 = l2w[n], w21 = l2w[n + 16];
    float l2bias = l2b[0];

    #pragma unroll
    for (int r = 0; r < 8; ++r) {
        float h0 = clamp01(c0[r] + bi0);
        float h1 = clamp01(c1[r] + bi1);
        float p  = h0 * w20 + h1 * w21;
        p += __shfl_xor(p, 1, 32);
        p += __shfl_xor(p, 2, 32);
        p += __shfl_xor(p, 4, 32);
        p += __shfl_xor(p, 8, 32);
        if (m == 0)
            out[rowbase + r + hi * 8] = p + l2bias;
    }
}

extern "C" void kernel_launch(void* const* d_in, const int* in_sizes, int n_in,
                              void* d_out, int out_size, void* d_ws, size_t ws_size,
                              hipStream_t stream) {
    const int*           widx = (const int*)d_in[0];
    const int*           woff = (const int*)d_in[1];
    const int*           bidx = (const int*)d_in[2];
    const int*           boff = (const int*)d_in[3];
    const unsigned char* stm  = (const unsigned char*)d_in[4];
    const float*         ftw  = (const float*)d_in[5];
    const float*         ftb  = (const float*)d_in[6];
    const float*         l1w  = (const float*)d_in[7];
    const float*         l1b  = (const float*)d_in[8];
    const float*         l2w  = (const float*)d_in[9];
    const float*         l2b  = (const float*)d_in[10];
    float*               out  = (float*)d_out;

    int nbags = in_sizes[1];   // B = 16384
    int wtot  = in_sizes[0];   // B*F
    int btot  = in_sizes[2];   // B*F

    _Float16* combined = (_Float16*)d_ws; // B x 512 f16 = 16 MB scratch

    dim3 blk(256);
    int threads1 = nbags * 32;               // one wave per bag
    dim3 grd1((threads1 + 255) / 256);
    nnue_embed_kernel<<<grd1, blk, 0, stream>>>(widx, woff, bidx, boff, stm,
                                                ftw, ftb, combined,
                                                nbags, wtot, btot);

    int tiles = (nbags + 15) / 16;           // one wave per 16-row tile
    dim3 grd2((tiles + 7) / 8);
    nnue_head_kernel<<<grd2, blk, 0, stream>>>(combined, l1w, l1b, l2w, l2b,
                                               out, nbags);
}
